// Self_Attention_55860344652429
// MI455X (gfx1250) — matrix operand order repaired and checked by reading, not compiled, not run on hardware
//
#include <hip/hip_runtime.h>
#include <hip/hip_bf16.h>
#include <math.h>

#define NEG_FILL (-1000000.0f)

// Problem constants from the reference
#define BS 128
#define SL 512
#define HD 1024
#define DD 512

typedef __attribute__((ext_vector_type(16))) _Float16 v16h;
typedef __attribute__((ext_vector_type(8)))  _Float16 v8h;
typedef __attribute__((ext_vector_type(4)))  _Float16 v4h;
typedef __attribute__((ext_vector_type(8)))  float    v8f;

// Branch-free tanh: 1 - 2/(exp(2a)+1).  exp->inf gives +1, exp->0 gives -1.
// Uses native v_exp_f32 + v_rcp_f32; ~1ulp of rcp error, plenty for softmax.
__device__ __forceinline__ float fast_tanh(float a) {
  float e = __expf(2.0f * a);
  return 1.0f - 2.0f * __builtin_amdgcn_rcpf(e + 1.0f);
}

// ---------------------------------------------------------------------------
// Kernel 0: convert Wq (fp32, [DD][HD] row-major) -> f16 workspace copy.
// ---------------------------------------------------------------------------
__global__ void cvt_wq_f16(const float* __restrict__ wq,
                           _Float16* __restrict__ wqh) {
  int i = (blockIdx.x * blockDim.x + threadIdx.x) * 4;  // DD*HD divisible by 4
  float4 v = *(const float4*)(wq + i);
  v4h h = { (_Float16)v.x, (_Float16)v.y, (_Float16)v.z, (_Float16)v.w };
  *(v4h*)(wqh + i) = h;
}

// ---------------------------------------------------------------------------
// Kernel 1: fused  score[row] = sum_d tanh( x[row,:]·Wq[d,:] + bq[d] ) * vq[d]
// Rows = flattened (b, s): 65536 rows, K = HD = 1024, N = DD = 512.
// WG = 64 threads (2 waves). Wave owns 32 rows = 2 M-tiles; 2 N-tiles per
// iteration -> 4 independent C chains, each B load reused across 2 M-tiles,
// each A load across 2 N-tiles.  x tile staged in LDS as f16 (132 KB ->
// 2 WGs/WGP).  WMMA: v_wmma_f32_16x16x32_f16.
// ---------------------------------------------------------------------------
#define MROWS 64
#define LDS_STRIDE 1032   // 1024 + 8 f16 pad -> bank-conflict-free b128 reads

__global__ void __launch_bounds__(64)
attn_score_gemm(const float* __restrict__ x,
                const _Float16* __restrict__ wqh,
                const float* __restrict__ bq,
                const float* __restrict__ vq,
                float* __restrict__ scores) {
  extern __shared__ _Float16 lds[];   // MROWS * LDS_STRIDE f16 = 132096 B

  const int tid   = threadIdx.x;      // 64 threads = 2 waves
  const int wave  = tid >> 5;
  const int lane  = tid & 31;
  const int lrow  = lane & 15;        // A-row / B-column / C-column lane id
  const int khalf = lane >> 4;        // which K half this lane holds
  const size_t mbase = (size_t)blockIdx.x * MROWS;   // first global row

  // --- stage x[mbase .. mbase+63][0..1023] into LDS as f16 (coalesced) ---
  {
    const float4* xs = (const float4*)(x + mbase * HD);
    #pragma unroll 4
    for (int i = tid; i < MROWS * HD / 4; i += 64) {
      float4 v = xs[i];
      int e   = i * 4;
      int row = e >> 10;          // /HD
      int col = e & (HD - 1);
      v4h h = { (_Float16)v.x, (_Float16)v.y, (_Float16)v.z, (_Float16)v.w };
      *(v4h*)(lds + row * LDS_STRIDE + col) = h;
    }
  }
  __syncthreads();

  // Wave owns rows [wave*32, wave*32+31]: two 16-row A tiles.
  const int arow0 = wave * 32 + lrow;
  const int arow1 = arow0 + 16;
  const _Float16* ap0 = lds + arow0 * LDS_STRIDE + khalf * 16;
  const _Float16* ap1 = lds + arow1 * LDS_STRIDE + khalf * 16;

  float sc0[8], sc1[8];
  #pragma unroll
  for (int r = 0; r < 8; ++r) { sc0[r] = 0.0f; sc1[r] = 0.0f; }

  // --- N loop over d columns, 2 N-tiles (32 cols) per iteration ---
  for (int nb = 0; nb < DD; nb += 32) {
    v8f c00 = {};  // (Mtile0, Ntile0)
    v8f c01 = {};  // (Mtile0, Ntile1)
    v8f c10 = {};  // (Mtile1, Ntile0)
    v8f c11 = {};  // (Mtile1, Ntile1)
    const int d0 = nb + lrow;
    const int d1 = nb + 16 + lrow;
    // B layout: lane = column, elements = 16 consecutive K at khalf*16
    const _Float16* b0p = wqh + (size_t)d0 * HD + khalf * 16;
    const _Float16* b1p = wqh + (size_t)d1 * HD + khalf * 16;

    #pragma unroll 4
    for (int kk = 0; kk < HD; kk += 32) {
      // A 16x32 f16: elems 0..15 = K[khalf*16 .. +15]
      v8h a0_lo = *(const v8h*)(ap0 + kk);
      v8h a0_hi = *(const v8h*)(ap0 + kk + 8);
      v8h a1_lo = *(const v8h*)(ap1 + kk);
      v8h a1_hi = *(const v8h*)(ap1 + kk + 8);
      v16h a0 = __builtin_shufflevector(a0_lo, a0_hi,
                                        0,1,2,3,4,5,6,7,8,9,10,11,12,13,14,15);
      v16h a1 = __builtin_shufflevector(a1_lo, a1_hi,
                                        0,1,2,3,4,5,6,7,8,9,10,11,12,13,14,15);
      v16h b0 = *(const v16h*)(b0p + kk);
      v16h b1 = *(const v16h*)(b1p + kk);
      c00 = __builtin_amdgcn_wmma_f32_16x16x32_f16(false, a0, false, b0,
                                                   (short)0, c00, false, false);
      c01 = __builtin_amdgcn_wmma_f32_16x16x32_f16(false, a0, false, b1,
                                                   (short)0, c01, false, false);
      c10 = __builtin_amdgcn_wmma_f32_16x16x32_f16(false, a1, false, b0,
                                                   (short)0, c10, false, false);
      c11 = __builtin_amdgcn_wmma_f32_16x16x32_f16(false, a1, false, b1,
                                                   (short)0, c11, false, false);
    }

    // Epilogue: branch-free tanh + vq-weighted partial reduction
    const float bq0 = bq[d0], vq0 = vq[d0];
    const float bq1 = bq[d1], vq1 = vq[d1];
    #pragma unroll
    for (int r = 0; r < 8; ++r) {
      sc0[r] += fast_tanh(c00[r] + bq0) * vq0;
      sc0[r] += fast_tanh(c01[r] + bq1) * vq1;
      sc1[r] += fast_tanh(c10[r] + bq0) * vq0;
      sc1[r] += fast_tanh(c11[r] + bq1) * vq1;
    }
  }

  // Reduce across the 16 N-lanes of each half-wave (C cols live per-lane)
  #pragma unroll
  for (int off = 8; off >= 1; off >>= 1) {
    #pragma unroll
    for (int r = 0; r < 8; ++r) {
      sc0[r] += __shfl_xor(sc0[r], off, 16);
      sc1[r] += __shfl_xor(sc1[r], off, 16);
    }
  }

  // C layout: VGPR r holds row M = r + 8*khalf -> lanes 0 and 16 hold results
  if (lrow == 0) {
    size_t base = mbase + (size_t)wave * 32 + khalf * 8;
    #pragma unroll
    for (int r = 0; r < 8; ++r) {
      scores[base + r]      = sc0[r];
      scores[base + 16 + r] = sc1[r];
    }
  }
}

// ---------------------------------------------------------------------------
// Kernel 2: per-batch masked softmax of 512 scores, then
//           res[b,h] = sum_s w[b,s] * x[b,s,h]   (coalesced float4 stream)
// ---------------------------------------------------------------------------
__global__ void __launch_bounds__(256)
softmax_weighted_sum(const float* __restrict__ x,
                     const unsigned char* __restrict__ mask,
                     const float* __restrict__ scores,
                     float* __restrict__ out) {
  const int b = blockIdx.x;
  const int tid = threadIdx.x;        // 256 threads

  __shared__ float w[SL];
  __shared__ float red[256];

  // load 2 scores/thread, apply mask
  float v0 = scores[b * SL + tid];
  float v1 = scores[b * SL + 256 + tid];
  if (mask[b * SL + tid])       v0 = NEG_FILL;
  if (mask[b * SL + 256 + tid]) v1 = NEG_FILL;

  // block max
  red[tid] = fmaxf(v0, v1);
  __syncthreads();
  for (int off = 128; off > 0; off >>= 1) {
    if (tid < off) red[tid] = fmaxf(red[tid], red[tid + off]);
    __syncthreads();
  }
  const float m = red[0];
  __syncthreads();

  // block sum of exp
  float e0 = expf(v0 - m);
  float e1 = expf(v1 - m);
  red[tid] = e0 + e1;
  __syncthreads();
  for (int off = 128; off > 0; off >>= 1) {
    if (tid < off) red[tid] += red[tid + off];
    __syncthreads();
  }
  const float inv = 1.0f / red[0];
  w[tid]       = e0 * inv;
  w[tid + 256] = e1 * inv;
  __syncthreads();

  // weighted sum over s: thread owns h = tid*4 .. tid*4+3  (float4 coalesced)
  const float4* x4 = (const float4*)(x + (size_t)b * SL * HD);
  float4 acc = make_float4(0.f, 0.f, 0.f, 0.f);
  for (int s = 0; s < SL; ++s) {
    const float ws = w[s];
    float4 xv = x4[(size_t)s * (HD / 4) + tid];
    acc.x += ws * xv.x;
    acc.y += ws * xv.y;
    acc.z += ws * xv.z;
    acc.w += ws * xv.w;
  }
  ((float4*)out)[(size_t)b * (HD / 4) + tid] = acc;
}

// ---------------------------------------------------------------------------
extern "C" void kernel_launch(void* const* d_in, const int* in_sizes, int n_in,
                              void* d_out, int out_size, void* d_ws, size_t ws_size,
                              hipStream_t stream) {
  const float*         x    = (const float*)d_in[0];
  const unsigned char* mask = (const unsigned char*)d_in[1];  // bool array
  const float*         Wq   = (const float*)d_in[2];
  const float*         bq   = (const float*)d_in[3];
  const float*         vq   = (const float*)d_in[4];
  float*               out  = (float*)d_out;

  // workspace layout: [0, 1MB) Wq in f16 ; then 65536 f32 scores
  _Float16* wqh    = (_Float16*)d_ws;
  float*    scores = (float*)((char*)d_ws + (size_t)DD * HD * sizeof(_Float16));

  // K0: Wq fp32 -> f16   (DD*HD = 524288 elems, 4/thread)
  cvt_wq_f16<<<(DD * HD) / (128 * 4), 128, 0, stream>>>(Wq, wqh);

  // K1: fused GEMM + tanh + vq reduction -> scores
  const int nblocks = (BS * SL) / MROWS;            // 1024
  const size_t lds_bytes = (size_t)MROWS * LDS_STRIDE * sizeof(_Float16);
  attn_score_gemm<<<nblocks, 64, lds_bytes, stream>>>(x, wqh, bq, vq, scores);

  // K2: masked softmax + weighted sum -> out [BS, HD]
  softmax_weighted_sum<<<BS, 256, 0, stream>>>(x, mask, scores, out);
}
